// Attention_87462714016262
// MI455X (gfx1250) — matrix operand order repaired and checked by reading, not compiled, not run on hardware
//
#include <hip/hip_runtime.h>
#include <hip/hip_bf16.h>

// ---------------------------------------------------------------------------
// MLA attention with sparse top-k index selection, CDNA5 (gfx1250) WMMA path.
// B=1, S=2048, DIM=2048, H=16, HD=128, DR=64, QLORA=768, KVLORA=512,
// HI=8, DI=64, TOPK=512.
//
// All matmuls run on v_wmma_f32_16x16x32_bf16 with f32 accumulation.
// GEMM tiles are staged global->LDS with the Tensor Data Mover
// (tensor_load_to_lds, double-buffered, TENSORcnt-pipelined) when the
// builtin is available; otherwise with b128 vector loads.
// ---------------------------------------------------------------------------

typedef __attribute__((ext_vector_type(16))) __bf16 v16bf;
typedef __attribute__((ext_vector_type(8)))  __bf16 v8bf;
typedef __attribute__((ext_vector_type(8)))  float  v8f;
typedef __attribute__((ext_vector_type(4)))  unsigned int v4u;
typedef __attribute__((ext_vector_type(4)))  int v4i;
typedef __attribute__((ext_vector_type(8)))  int v8i;

#if __has_builtin(__builtin_amdgcn_tensor_load_to_lds) && \
    __has_builtin(__builtin_amdgcn_s_wait_tensorcnt)
#define HAVE_TDM 1
#else
#define HAVE_TDM 0
#endif

#define S_LEN   2048
#define DIM     2048
#define NHEAD   16
#define HDIM    128
#define DRDIM   64
#define QLORA_  768
#define KVLORA_ 512
#define HI_     8
#define DI_     64
#define TOPK_   512
#define NEGV    (-1.0e30f)

__device__ __forceinline__ __bf16 f2bf(float f) {
    union { float f; unsigned u; } in; in.f = f;
    unsigned r = in.u + 0x7FFFu + ((in.u >> 16) & 1u);   // RNE
    union { unsigned short s; __bf16 b; } out;
    out.s = (unsigned short)(r >> 16);
    return out.b;
}
__device__ __forceinline__ float bf2f(__bf16 b) {
    union { __bf16 b; unsigned short s; } in; in.b = b;
    union { unsigned u; float f; } out; out.u = ((unsigned)in.s) << 16;
    return out.f;
}
__device__ __forceinline__ v8f v8f_zero() {
    v8f z = {0.f, 0.f, 0.f, 0.f, 0.f, 0.f, 0.f, 0.f};
    return z;
}

// A-fragment (16x32, 16-bit): element e -> K = (e>>3)*16 + ((lane&16)>>1) + (e&7).
__device__ __forceinline__ v16bf load_afrag(const __bf16* base, int lane) {
    union { v16bf v; v8bf h[2]; } u;
    int hb = (lane & 16) >> 1;
    u.h[0] = *(const v8bf*)(base + hb);
    u.h[1] = *(const v8bf*)(base + 16 + hb);
    return u.v;
}
// B-fragment (32x16, 16-bit, operand stored N-major): element e -> K = (e>>3)*16 + ((lane&16)>>1) + (e&7).
__device__ __forceinline__ v16bf load_bfrag(const __bf16* base, int lane) {
    union { v16bf v; v8bf h[2]; } u;
    int kb = (lane & 16) >> 1;
    u.h[0] = *(const v8bf*)(base + kb);
    u.h[1] = *(const v8bf*)(base + 16 + kb);
    return u.v;
}

#if HAVE_TDM
// ---------------------------------------------------------------------------
// Issue one 2D TENSOR_LOAD_TO_LDS: copy a tile_d1 x tile_d0 tile of bf16
// (row stride stride0 elements) from gaddr into LDS at lds_addr, packed
// row-major tile_d0 wide.  Rows/cols past tensor_d1/tensor_d0 read zero.
// D# packing per CDNA5 ISA 08_async_tensor 8.3/8.4.
// ---------------------------------------------------------------------------
__device__ __forceinline__ void tdm_load_2d(unsigned lds_addr, const void* gaddr,
                                            unsigned tensor_d0, unsigned tensor_d1,
                                            unsigned tile_d0, unsigned tile_d1,
                                            unsigned long long stride0)
{
    unsigned long long ga = (unsigned long long)(size_t)gaddr;
    v4u g0;
    g0[0] = 1u;                                            // count=1 (user D#)
    g0[1] = lds_addr;                                      // LDS byte address
    g0[2] = (unsigned)(ga & 0xFFFFFFFFu);                  // global_addr[31:0]
    g0[3] = (unsigned)((ga >> 32) & 0x01FFFFFFu) | (2u << 30);  // [56:32]|type=2
    v8i g1;
    g1[0] = (int)(1u << 16);                               // data_size=2B
    g1[1] = (int)((tensor_d0 & 0xFFFFu) << 16);            // tensor_dim0[15:0]
    g1[2] = (int)((tensor_d0 >> 16) | ((tensor_d1 & 0xFFFFu) << 16));
    g1[3] = (int)((tensor_d1 >> 16) | (tile_d0 << 16));    // tile_dim0
    g1[4] = (int)(tile_d1 & 0xFFFFu);                      // tile_dim1 (dim2=0)
    g1[5] = (int)(stride0 & 0xFFFFFFFFu);                  // dim0 stride
    g1[6] = (int)((stride0 >> 32) & 0xFFFFu);
    g1[7] = 0;
    v4i z4 = {0, 0, 0, 0};
#if defined(__clang_major__) && (__clang_major__ >= 23)
    v8i z8 = {0, 0, 0, 0, 0, 0, 0, 0};
    __builtin_amdgcn_tensor_load_to_lds(g0, g1, z4, z4, z8, 0);
#else
    __builtin_amdgcn_tensor_load_to_lds(g0, g1, z4, z4, 0);
#endif
}
#endif

// ---------------------------------------------------------------------------
// Flat f32 -> bf16 conversion.
// ---------------------------------------------------------------------------
__global__ __launch_bounds__(256)
void conv_bf16(const float* __restrict__ in, __bf16* __restrict__ out, int n)
{
    int i = blockIdx.x * 1024 + threadIdx.x;
#pragma unroll
    for (int j = 0; j < 4; ++j) {
        int idx = i + j * 256;
        if (idx < n) out[idx] = f2bf(in[idx]);
    }
}

// ---------------------------------------------------------------------------
// Tiled transpose + f32->bf16: in R x C (row-major) -> out C x R (row-major).
// ---------------------------------------------------------------------------
__global__ __launch_bounds__(256)
void transpose_bf16(const float* __restrict__ in, __bf16* __restrict__ out,
                    int R, int C)
{
    __shared__ float tile[32][33];
    int bx = blockIdx.x * 32;
    int by = blockIdx.y * 32;
    int tx = threadIdx.x & 31, ty = threadIdx.x >> 5;
#pragma unroll
    for (int i = 0; i < 32; i += 8) {
        int r = by + ty + i, c = bx + tx;
        tile[ty + i][tx] = (r < R && c < C) ? in[(size_t)r * C + c] : 0.f;
    }
    __syncthreads();
#pragma unroll
    for (int i = 0; i < 32; i += 8) {
        int orow = bx + ty + i;
        int ocol = by + tx;
        if (orow < C && ocol < R)
            out[(size_t)orow * R + ocol] = f2bf(tile[tx][ty + i]);
    }
}

// ---------------------------------------------------------------------------
// V transpose: kvb_bf[t][h*256+128+f] -> vT[(h*128+f)][t]   (bf16 -> bf16)
// ---------------------------------------------------------------------------
__global__ __launch_bounds__(256)
void transpose_v(const __bf16* __restrict__ kvb_bf, __bf16* __restrict__ vT)
{
    int gid = blockIdx.x * 256 + threadIdx.x;
    int t   = gid & (S_LEN - 1);
    int row = gid >> 11;                          // h*128 + f
    int h = row >> 7, f = row & 127;
    vT[gid] = kvb_bf[(size_t)t * (NHEAD * 256) + h * 256 + 128 + f];
}

// ---------------------------------------------------------------------------
// Tiled GEMM: C[M,N] = A[M,K] @ B[K,N].  A: bf16 row-major.  BT: bf16,
// B transposed (N x K row-major).  C: f32 or bf16 (c_bf16 flag).
// Block = 256 threads = 8 waves; block tile 64x128; wave tile 32x32.
// K % 32 == 0, N % 8 == 0, M % 64 == 0 at every call site.
// ---------------------------------------------------------------------------
__global__ __launch_bounds__(256)
void gemm_wmma(const __bf16* __restrict__ A, int lda,
               const __bf16* __restrict__ BT, int ldbt,
               void* __restrict__ C, int ldc,
               int M, int N, int K, int c_bf16)
{
    __shared__ __align__(16) __bf16 As[1 + HAVE_TDM][64][32];
    __shared__ __align__(16) __bf16 BsT[1 + HAVE_TDM][128][32];

    const int tid  = threadIdx.x;
    const int lane = tid & 31;
    const int wave = tid >> 5;
    const int wr   = wave & 1;
    const int wc   = wave >> 1;
    const int m0   = blockIdx.y * 64;
    const int n0   = blockIdx.x * 128;
    const int nl   = lane & 15;
    const int half = (lane >> 4) & 1;

    v8f acc[2][2];
#pragma unroll
    for (int i = 0; i < 2; ++i)
#pragma unroll
        for (int j = 0; j < 2; ++j) acc[i][j] = v8f_zero();

#if HAVE_TDM
    // Prime the pipeline: TDM the k0=0 tiles into buffer 0.
    if (wave == 0) {
        tdm_load_2d((unsigned)(size_t)&As[0][0][0],
                    &A[(size_t)m0 * lda], (unsigned)K, (unsigned)(M - m0),
                    32u, 64u, (unsigned long long)lda);
        tdm_load_2d((unsigned)(size_t)&BsT[0][0][0],
                    &BT[(size_t)n0 * ldbt], (unsigned)K, (unsigned)(N - n0),
                    32u, 128u, (unsigned long long)ldbt);
    }
#endif

    int buf = 0;
    for (int k0 = 0; k0 < K; k0 += 32) {
#if HAVE_TDM
        if (wave == 0) {
            if (k0 + 32 < K) {
                // Prefetch next K-tiles into the other buffer while this
                // buffer's pair finishes; overlap TDM with WMMA below.
                tdm_load_2d((unsigned)(size_t)&As[buf ^ 1][0][0],
                            &A[(size_t)m0 * lda + k0 + 32],
                            (unsigned)(K - k0 - 32), (unsigned)(M - m0),
                            32u, 64u, (unsigned long long)lda);
                tdm_load_2d((unsigned)(size_t)&BsT[buf ^ 1][0][0],
                            &BT[(size_t)n0 * ldbt + k0 + 32],
                            (unsigned)(K - k0 - 32), (unsigned)(N - n0),
                            32u, 128u, (unsigned long long)ldbt);
                __builtin_amdgcn_s_wait_tensorcnt(2);  // current pair done
            } else {
                __builtin_amdgcn_s_wait_tensorcnt(0);
            }
        }
        __syncthreads();
#else
        // Fallback staging: b128 vector loads, clamped (no branches; rows
        // past N only feed discarded C columns).
        {
            int r = tid >> 2, c8 = (tid & 3) * 8;
            *(v8bf*)&As[0][r][c8] =
                *(const v8bf*)&A[(size_t)(m0 + r) * lda + k0 + c8];
        }
#pragma unroll
        for (int i = 0; i < 2; ++i) {
            int chunk = tid + i * 256;
            int n = chunk >> 2, c8 = (chunk & 3) * 8;
            int nn = min(n0 + n, N - 8);
            *(v8bf*)&BsT[0][n][c8] =
                *(const v8bf*)&BT[(size_t)nn * ldbt + k0 + c8];
        }
        __syncthreads();
#endif

        v16bf af[2], bfr[2];
#pragma unroll
        for (int t = 0; t < 2; ++t) {
            af[t]  = load_afrag(&As[buf][wr * 32 + t * 16 + nl][0], lane);
            bfr[t] = load_bfrag(&BsT[buf][wc * 32 + t * 16 + nl][0], lane);
        }
#pragma unroll
        for (int i = 0; i < 2; ++i)
#pragma unroll
            for (int j = 0; j < 2; ++j)
                acc[i][j] = __builtin_amdgcn_wmma_f32_16x16x32_bf16(
                    false, af[i], false, bfr[j], (short)0, acc[i][j], false, false);
        __syncthreads();
        buf ^= HAVE_TDM;
    }

#pragma unroll
    for (int i = 0; i < 2; ++i)
#pragma unroll
        for (int j = 0; j < 2; ++j)
#pragma unroll
            for (int r = 0; r < 8; ++r) {
                int gm = m0 + wr * 32 + i * 16 + r + 8 * half;
                int gn = n0 + wc * 32 + j * 16 + nl;
                if (gm < M && gn < N) {
                    if (c_bf16)
                        ((__bf16*)C)[(size_t)gm * ldc + gn] = f2bf(acc[i][j][r]);
                    else
                        ((float*)C)[(size_t)gm * ldc + gn] = acc[i][j][r];
                }
            }
}

// ---------------------------------------------------------------------------
// Rope (pairwise) + optional 64-wide FWHT on a bf16 buffer, one wave per
// (row, head).  dim == 64 everywhere: lane L owns elements 2L, 2L+1.
// ---------------------------------------------------------------------------
__global__ __launch_bounds__(256)
void rope_fwht_bf(__bf16* __restrict__ buf,
                  const float* __restrict__ cosb, const float* __restrict__ sinb,
                  int S, int heads, int rowStride, int headOffset, int headStride,
                  int doFwht)
{
    int gw   = blockIdx.x * 8 + (threadIdx.x >> 5);
    int lane = threadIdx.x & 31;
    if (gw >= S * heads) return;          // wave-uniform
    int s = gw / heads, h = gw % heads;
    __bf16* p = buf + (size_t)s * rowStride + headOffset + h * headStride;

    float x1 = bf2f(p[2 * lane]), x2 = bf2f(p[2 * lane + 1]);
    float c  = cosb[s * 32 + lane], sn = sinb[s * 32 + lane];
    float a  = x1 * c - x2 * sn;
    float b  = x1 * sn + x2 * c;

    if (doFwht) {
        float na = a + b, nb = a - b;     // distance-1 butterfly (in lane)
        a = na; b = nb;
#pragma unroll
        for (int bit = 0; bit < 5; ++bit) {
            int m = 1 << bit;
            float oa = __shfl_xor(a, m, 32);
            float ob = __shfl_xor(b, m, 32);
            bool hi = (lane & m) != 0;
            a = hi ? (oa - a) : (a + oa);
            b = hi ? (ob - b) : (b + ob);
        }
        a *= 0.125f; b *= 0.125f;         // 64^-0.5
    }
    p[2 * lane] = f2bf(a); p[2 * lane + 1] = f2bf(b);
}

// ---------------------------------------------------------------------------
// idx_scores[s,t] = DI^-0.5 * sum_h relu(qi[s,h,:].ki[t,:]) * gate[s,h],
// causal else NEG.  One wave per 16x16 (s,t) tile; 2 WMMAs per index head.
// ---------------------------------------------------------------------------
__global__ __launch_bounds__(256)
void idx_logits(const __bf16* __restrict__ qi,   // S x 512 bf16
                const __bf16* __restrict__ ki,   // S x 64 bf16
                const float* __restrict__ gate,  // S x 8 f32
                float* __restrict__ scores)      // S x S f32
{
    int gw   = blockIdx.x * 8 + (threadIdx.x >> 5);
    int lane = threadIdx.x & 31;
    int s0 = (gw >> 7) * 16, t0 = (gw & 127) * 16;
    int nl = lane & 15, half = (lane >> 4) & 1;

    if (t0 > s0 + 15) {                   // fully non-causal tile
#pragma unroll
        for (int r = 0; r < 8; ++r)
            scores[(size_t)(s0 + r + 8 * half) * S_LEN + t0 + nl] = NEGV;
        return;
    }

    float total[8];
#pragma unroll
    for (int r = 0; r < 8; ++r) total[r] = 0.f;

    for (int h = 0; h < HI_; ++h) {
        v8f cacc = v8f_zero();
#pragma unroll
        for (int kk = 0; kk < 2; ++kk) {
            v16bf aq = load_afrag(
                qi + (size_t)(s0 + nl) * (HI_ * DI_) + h * DI_ + kk * 32, lane);
            v16bf bk = load_bfrag(
                ki + (size_t)(t0 + nl) * DI_ + kk * 32, lane);
            cacc = __builtin_amdgcn_wmma_f32_16x16x32_bf16(
                false, aq, false, bk, (short)0, cacc, false, false);
        }
#pragma unroll
        for (int r = 0; r < 8; ++r) {
            int s = s0 + r + 8 * half;
            total[r] += fmaxf(cacc[r], 0.f) * gate[s * HI_ + h];
        }
    }

    int t = t0 + nl;
#pragma unroll
    for (int r = 0; r < 8; ++r) {
        int s = s0 + r + 8 * half;
        scores[(size_t)s * S_LEN + t] = (t <= s) ? total[r] * 0.125f : NEGV;
    }
}

// ---------------------------------------------------------------------------
// Exact k-th largest per row via 8x4-bit radix select on monotone float keys.
// ---------------------------------------------------------------------------
__device__ __forceinline__ unsigned mono(float f) {
    unsigned u = __float_as_uint(f);
    return (u & 0x80000000u) ? ~u : (u | 0x80000000u);
}
__device__ __forceinline__ float unmono(unsigned k) {
    unsigned u = (k & 0x80000000u) ? (k ^ 0x80000000u) : ~k;
    return __uint_as_float(u);
}

__global__ __launch_bounds__(256)
void topk_select(const float* __restrict__ scores, float* __restrict__ kth)
{
    __shared__ unsigned cnt[16];
    const float* rp = scores + (size_t)blockIdx.x * S_LEN;
    int tid = threadIdx.x;
    unsigned prefix = 0;
    int remaining = TOPK_;

    for (int d = 7; d >= 0; --d) {
        if (tid < 16) cnt[tid] = 0;
        __syncthreads();
        int shift = d * 4;
        unsigned mask_hi = (d == 7) ? 0u : (0xFFFFFFFFu << (shift + 4));
        for (int i = tid; i < S_LEN; i += 256) {
            unsigned key = mono(rp[i]);
            if ((key & mask_hi) == (prefix & mask_hi))
                atomicAdd(&cnt[(key >> shift) & 15], 1u);
        }
        __syncthreads();
        unsigned cum = 0; int chosen = 0;
        for (int dig = 15; dig >= 0; --dig) {
            unsigned c = cnt[dig];
            if (cum + c >= (unsigned)remaining) { chosen = dig; break; }
            cum += c;
        }
        remaining -= (int)cum;
        prefix |= ((unsigned)chosen) << shift;
        __syncthreads();
    }
    if (tid == 0) kth[blockIdx.x] = unmono(prefix);
}

// ---------------------------------------------------------------------------
// Flash attention with top-k + causal mask.  One wave per (head, 16-q tile).
// Scores: 6 WMMAs per 16-key tile; P.V: 8 WMMAs per 32-key step (V read
// from feature-major vT so all fragments are contiguous b128 loads).
// ---------------------------------------------------------------------------
__global__ __launch_bounds__(256)
void mla_flash(const __bf16* __restrict__ q,      // S x 3072 bf16
               const __bf16* __restrict__ kvb,    // S x 4096 bf16 (k_nope|v)
               const __bf16* __restrict__ kpe,    // S x 576 bf16 (k_pe at +512)
               const __bf16* __restrict__ vT,     // (H*HD) x S bf16
               const float* __restrict__ scores,  // S x S f32
               const float* __restrict__ kth,     // S f32
               __bf16* __restrict__ attn_out)     // S x 2048 bf16
{
    __shared__ __align__(16) __bf16 Ps[8][16][32];
    int wv   = threadIdx.x >> 5;
    int lane = threadIdx.x & 31;
    int gw   = blockIdx.x * 8 + wv;
    int h    = gw >> 7;
    int s0   = (gw & 127) * 16;
    int nl   = lane & 15, half = (lane >> 4) & 1;
    const float scale = 0.0721687836487032f;   // (HD+DR)^-0.5

    // Q fragments: 6 K-chunks of 32 covering features 0..191.
    v16bf qf[6];
    const __bf16* qrow = q + (size_t)(s0 + nl) * (NHEAD * 192) + h * 192;
#pragma unroll
    for (int c = 0; c < 6; ++c) qf[c] = load_afrag(qrow + c * 32, lane);

    v8f O[8];
#pragma unroll
    for (int nf = 0; nf < 8; ++nf) O[nf] = v8f_zero();
    float m[8], l[8], kthv[8];
#pragma unroll
    for (int r = 0; r < 8; ++r) {
        m[r] = NEGV; l[r] = 0.f;
        kthv[r] = kth[s0 + r + 8 * half];
    }

    const int shi = s0 + 15;
    for (int t0 = 0; t0 <= shi; t0 += 32) {
        float sv[2][8];
#pragma unroll
        for (int sub = 0; sub < 2; ++sub) {
            int tb = t0 + sub * 16;
            if (tb <= shi) {
                int t = tb + nl;
                v8f cacc = v8f_zero();
#pragma unroll
                for (int ch = 0; ch < 6; ++ch) {
                    const __bf16* kb = (ch < 4)
                        ? kvb + (size_t)t * (NHEAD * 256) + h * 256 + ch * 32
                        : kpe + (size_t)t * (KVLORA_ + DRDIM) + KVLORA_ + (ch - 4) * 32;
                    v16bf bfr = load_bfrag(kb, lane);
                    cacc = __builtin_amdgcn_wmma_f32_16x16x32_bf16(
                        false, qf[ch], false, bfr, (short)0, cacc, false, false);
                }
#pragma unroll
                for (int r = 0; r < 8; ++r) {
                    int s = s0 + r + 8 * half;
                    bool ok = (t <= s) &&
                              (scores[(size_t)s * S_LEN + t] >= kthv[r]);
                    sv[sub][r] = ok ? cacc[r] * scale : NEGV;
                }
            } else {
#pragma unroll
                for (int r = 0; r < 8; ++r) sv[sub][r] = NEGV;
            }
        }

        // Online softmax (per-row stats live in each 16-lane half).
        float alpha[8];
#pragma unroll
        for (int r = 0; r < 8; ++r) {
            float mx = fmaxf(sv[0][r], sv[1][r]);
#pragma unroll
            for (int b = 0; b < 4; ++b) mx = fmaxf(mx, __shfl_xor(mx, 1 << b, 32));
            float mnew = fmaxf(m[r], mx);
            alpha[r] = __expf(m[r] - mnew);
            float p0 = (sv[0][r] > -1e29f) ? __expf(sv[0][r] - mnew) : 0.f;
            float p1 = (sv[1][r] > -1e29f) ? __expf(sv[1][r] - mnew) : 0.f;
            sv[0][r] = p0; sv[1][r] = p1;
            float rs = p0 + p1;
#pragma unroll
            for (int b = 0; b < 4; ++b) rs += __shfl_xor(rs, 1 << b, 32);
            l[r] = l[r] * alpha[r] + rs;
            m[r] = mnew;
        }
#pragma unroll
        for (int nf = 0; nf < 8; ++nf)
#pragma unroll
            for (int r = 0; r < 8; ++r) O[nf][r] *= alpha[r];

        // Stage P (C layout) into LDS, re-read as A fragment (16x32).
#pragma unroll
        for (int r = 0; r < 8; ++r) {
            int M = r + 8 * half;
            Ps[wv][M][nl]      = f2bf(sv[0][r]);
            Ps[wv][M][16 + nl] = f2bf(sv[1][r]);
        }
        asm volatile("s_wait_dscnt 0" ::: "memory");
        v16bf pa = load_afrag(&Ps[wv][nl][0], lane);

        // O += P @ V   (vT rows are features; K = keys, contiguous)
#pragma unroll
        for (int nf = 0; nf < 8; ++nf) {
            v16bf vf = load_bfrag(
                vT + (size_t)(h * HDIM + nf * 16 + nl) * S_LEN + t0, lane);
            O[nf] = __builtin_amdgcn_wmma_f32_16x16x32_bf16(
                false, pa, false, vf, (short)0, O[nf], false, false);
        }
    }

#pragma unroll
    for (int nf = 0; nf < 8; ++nf)
#pragma unroll
        for (int r = 0; r < 8; ++r) {
            int s = s0 + r + 8 * half;
            attn_out[(size_t)s * (NHEAD * HDIM) + h * HDIM + nf * 16 + nl] =
                f2bf(O[nf][r] / l[r]);
        }
}

// ---------------------------------------------------------------------------
// Host-side orchestration.
// ---------------------------------------------------------------------------
extern "C" void kernel_launch(void* const* d_in, const int* in_sizes, int n_in,
                              void* d_out, int out_size, void* d_ws, size_t ws_size,
                              hipStream_t stream)
{
    (void)in_sizes; (void)n_in; (void)out_size; (void)ws_size;
    const float* x      = (const float*)d_in[0];
    const float* fcos   = (const float*)d_in[1];
    const float* fsin   = (const float*)d_in[2];
    const float* wq_a   = (const float*)d_in[3];
    const float* wq_b   = (const float*)d_in[4];
    const float* wkv_a  = (const float*)d_in[5];
    const float* wkv_b  = (const float*)d_in[6];
    const float* wo     = (const float*)d_in[7];
    const float* wq_idx = (const float*)d_in[8];
    const float* wk_idx = (const float*)d_in[9];
    const float* w_gate = (const float*)d_in[10];

    char* wp = (char*)d_ws;
    auto alloc = [&](size_t bytes) -> char* {
        char* p = wp; wp += (bytes + 255) & ~(size_t)255; return p;
    };
    __bf16* xbf    = (__bf16*)alloc((size_t)S_LEN * DIM * 2);
    __bf16* wqiT   = (__bf16*)alloc((size_t)(HI_ * DI_) * DIM * 2);
    __bf16* wkiT   = (__bf16*)alloc((size_t)DI_ * DIM * 2);
    __bf16* wgT    = (__bf16*)alloc((size_t)HI_ * DIM * 2);
    __bf16* wqaT   = (__bf16*)alloc((size_t)QLORA_ * DIM * 2);
    __bf16* wqbT   = (__bf16*)alloc((size_t)(NHEAD * 192) * QLORA_ * 2);
    __bf16* wkvaT  = (__bf16*)alloc((size_t)(KVLORA_ + DRDIM) * DIM * 2);
    __bf16* wkvbT  = (__bf16*)alloc((size_t)(NHEAD * 256) * KVLORA_ * 2);
    __bf16* woT    = (__bf16*)alloc((size_t)DIM * (NHEAD * HDIM) * 2);
    __bf16* qi_bf  = (__bf16*)alloc((size_t)S_LEN * (HI_ * DI_) * 2);
    __bf16* ki_bf  = (__bf16*)alloc((size_t)S_LEN * DI_ * 2);
    float*  gate   = (float*)alloc((size_t)S_LEN * HI_ * 4);
    float*  sc     = (float*)alloc((size_t)S_LEN * S_LEN * 4);
    float*  kthb   = (float*)alloc((size_t)S_LEN * 4);
    __bf16* xa_bf  = (__bf16*)alloc((size_t)S_LEN * QLORA_ * 2);
    __bf16* q_bf   = (__bf16*)alloc((size_t)S_LEN * (NHEAD * 192) * 2);
    __bf16* kv_bf  = (__bf16*)alloc((size_t)S_LEN * (KVLORA_ + DRDIM) * 2);
    __bf16* kvb_bf = (__bf16*)alloc((size_t)S_LEN * (NHEAD * 256) * 2);
    __bf16* vT     = (__bf16*)alloc((size_t)(NHEAD * HDIM) * S_LEN * 2);
    __bf16* ao_bf  = (__bf16*)alloc((size_t)S_LEN * (NHEAD * HDIM) * 2);
    float*  out    = (float*)d_out;

    dim3 blk(256);
    auto gemm = [&](const __bf16* A, int lda, const __bf16* BT, int ldbt,
                    void* C, int ldc, int M, int N, int K, int c_bf16) {
        dim3 grid((N + 127) / 128, (M + 63) / 64);
        gemm_wmma<<<grid, blk, 0, stream>>>(A, lda, BT, ldbt, C, ldc, M, N, K, c_bf16);
    };
    auto transpose = [&](const float* in, __bf16* outb, int R, int Cc) {
        dim3 grid((Cc + 31) / 32, (R + 31) / 32);
        transpose_bf16<<<grid, blk, 0, stream>>>(in, outb, R, Cc);
    };

    // One-time conversions (x + all weights, weights transposed to N-major).
    conv_bf16<<<(S_LEN * DIM + 1023) / 1024, blk, 0, stream>>>(x, xbf, S_LEN * DIM);
    transpose(wq_idx, wqiT, DIM, HI_ * DI_);
    transpose(wk_idx, wkiT, DIM, DI_);
    transpose(w_gate, wgT,  DIM, HI_);
    transpose(wq_a,   wqaT, DIM, QLORA_);
    transpose(wq_b,   wqbT, QLORA_, NHEAD * 192);
    transpose(wkv_a,  wkvaT, DIM, KVLORA_ + DRDIM);
    transpose(wkv_b,  wkvbT, KVLORA_, NHEAD * 256);
    transpose(wo,     woT,  NHEAD * HDIM, DIM);

    // Index path.
    gemm(xbf, DIM, wqiT, DIM, qi_bf, HI_ * DI_, S_LEN, HI_ * DI_, DIM, 1);
    gemm(xbf, DIM, wkiT, DIM, ki_bf, DI_, S_LEN, DI_, DIM, 1);
    gemm(xbf, DIM, wgT,  DIM, gate,  HI_, S_LEN, HI_, DIM, 0);
    rope_fwht_bf<<<(S_LEN * HI_) / 8, blk, 0, stream>>>(qi_bf, fcos, fsin, S_LEN, HI_,
                                                        HI_ * DI_, 0, DI_, 1);
    rope_fwht_bf<<<S_LEN / 8, blk, 0, stream>>>(ki_bf, fcos, fsin, S_LEN, 1,
                                                DI_, 0, 0, 1);
    idx_logits<<<(S_LEN / 16) * (S_LEN / 16) / 8, blk, 0, stream>>>(qi_bf, ki_bf,
                                                                    gate, sc);
    topk_select<<<S_LEN, blk, 0, stream>>>(sc, kthb);

    // Q path.
    gemm(xbf, DIM, wqaT, DIM, xa_bf, QLORA_, S_LEN, QLORA_, DIM, 1);
    gemm(xa_bf, QLORA_, wqbT, QLORA_, q_bf, NHEAD * 192, S_LEN, NHEAD * 192, QLORA_, 1);
    rope_fwht_bf<<<(S_LEN * NHEAD) / 8, blk, 0, stream>>>(q_bf, fcos, fsin, S_LEN, NHEAD,
                                                          NHEAD * 192, HDIM, 192, 0);

    // KV path.
    gemm(xbf, DIM, wkvaT, DIM, kv_bf, KVLORA_ + DRDIM, S_LEN, KVLORA_ + DRDIM, DIM, 1);
    rope_fwht_bf<<<S_LEN / 8, blk, 0, stream>>>(kv_bf, fcos, fsin, S_LEN, 1,
                                                KVLORA_ + DRDIM, KVLORA_, 0, 0);
    gemm(kv_bf, KVLORA_ + DRDIM, wkvbT, KVLORA_, kvb_bf, NHEAD * 256,
         S_LEN, NHEAD * 256, KVLORA_, 1);
    transpose_v<<<(S_LEN * NHEAD * HDIM) / 256, blk, 0, stream>>>(kvb_bf, vT);

    // Attention + output projection.
    mla_flash<<<(NHEAD * (S_LEN / 16)) / 8, blk, 0, stream>>>(q_bf, kvb_bf, kv_bf,
                                                              vT, sc, kthb, ao_bf);
    gemm(ao_bf, NHEAD * HDIM, woT, NHEAD * HDIM, out, DIM,
         S_LEN, DIM, NHEAD * HDIM, 0);
}
